// MultiHeadAttn_49357764166084
// MI455X (gfx1250) — matrix operand order repaired and checked by reading, not compiled, not run on hardware
//
#include <hip/hip_runtime.h>
#include <hip/hip_bf16.h>
#include <cstddef>

// ---------------------------------------------------------------------------
// Problem constants (match reference)
// ---------------------------------------------------------------------------
#define SEQ     2048
#define BATCH   2
#define DMODEL  1024
#define NHEAD   16
#define DHEAD   64
#define ROWS    (SEQ * BATCH)          // 4096 flattened (t,b) rows
#define SCALE_F 0.125f                 // 1/sqrt(64)
#define MASK_VAL (-3.402823466e38f)    // jnp.finfo(f32).min

// ---------------------------------------------------------------------------
// CDNA5 WMMA types
// ---------------------------------------------------------------------------
typedef __bf16 v16bf __attribute__((ext_vector_type(16)));
typedef float  v8f   __attribute__((ext_vector_type(8)));
typedef float  f32x4 __attribute__((ext_vector_type(4)));   // native vec for NT stores

union FragBF {                 // 16 bf16 = one WMMA A or B operand (wave32)
    unsigned u[8];
    uint4    q[2];
    v16bf    v;
};

__device__ __forceinline__ v8f bf16_wmma(const FragBF& a, const FragBF& b, v8f c) {
    // D = A(16x32 bf16) * B(32x16 bf16) + C(16x16 f32)
    return __builtin_amdgcn_wmma_f32_16x16x32_bf16(
        /*neg_a=*/false, a.v, /*neg_b=*/false, b.v,
        /*c_mod=*/(short)0, c, /*reuse_a=*/false, /*reuse_b=*/false);
}

__device__ __forceinline__ unsigned short f32_to_bf16(float f) {
    unsigned u = __float_as_uint(f);
    unsigned r = u + 0x7FFFu + ((u >> 16) & 1u);   // round-to-nearest-even
    return (unsigned short)(r >> 16);
}

// ---------------------------------------------------------------------------
// Tensor Data Mover: 2D tile DMA global -> LDS (bf16 elements).
// D# bitfields per CDNA5 ISA.
// This toolchain exposes the 6-arg builtin:
//   (uint32x4 g0, int32x8 g1, int32x4 g2, int32x4 g3, int32x8 xtra, i32 cpol)
// ---------------------------------------------------------------------------
#if defined(__has_builtin)
#if __has_builtin(__builtin_amdgcn_tensor_load_to_lds) && \
    __has_builtin(__builtin_amdgcn_s_wait_tensorcnt)
#define HAVE_TDM 1
#endif
#endif

#ifdef HAVE_TDM
typedef unsigned int u32x4 __attribute__((ext_vector_type(4)));
typedef int          i32x8 __attribute__((ext_vector_type(8)));
typedef int          i32x4 __attribute__((ext_vector_type(4)));

__device__ __forceinline__ void tdm_load_tile_bf16(
    void* lds_dst, const void* gsrc,
    unsigned rows, unsigned row_elems, unsigned row_stride_elems) {
    unsigned long long ga = (unsigned long long)(size_t)gsrc;   // flat VA
    unsigned lds_off = (unsigned)(size_t)lds_dst;               // LDS byte offset

    u32x4 g0;
    g0[0] = 1u;                                      // count=1, user descriptor
    g0[1] = lds_off;                                 // lds_addr [63:32]
    g0[2] = (unsigned)ga;                            // global_addr [95:64]
    g0[3] = (unsigned)((ga >> 32) & 0x01FFFFFFu)     // global_addr [120:96]
          | (2u << 30);                              // type=2 ("image") [127:126]

    i32x8 g1;
    g1[0] = 0x10000;                                 // data_size=1 (2B), mask=0
    g1[1] = (int)(row_elems << 16);                  // tensor_dim0 low16 @ [63:48]
    g1[2] = (int)(rows << 16);                       // tensor_dim1 low16 @ [95:80]
    g1[3] = (int)(row_elems << 16);                  // tile_dim0 @ [127:112]
    g1[4] = (int)rows;                               // tile_dim1 @ [143:128]
    g1[5] = (int)row_stride_elems;                   // tensor_dim0_stride low32
    g1[6] = 0;
    g1[7] = 0;

    i32x4 z4 = {0, 0, 0, 0};                         // groups 2/3 unused (2D)
    i32x8 z8 = {0, 0, 0, 0, 0, 0, 0, 0};
    __builtin_amdgcn_tensor_load_to_lds(g0, g1, z4, z4, z8, 0);
}
#define TDM_WAIT() __builtin_amdgcn_s_wait_tensorcnt(0)
#else
#define TDM_WAIT() ((void)0)
#endif

// ---------------------------------------------------------------------------
// Kernel 1: f32 -> bf16 cast (vectorized x4)
// ---------------------------------------------------------------------------
__global__ void cast_bf16_kernel(const float* __restrict__ src,
                                 unsigned short* __restrict__ dst, int n4) {
    int i = blockIdx.x * blockDim.x + threadIdx.x;
    if (i < n4) {
        float4 f = ((const float4*)src)[i];
        uint2 p;
        p.x = (unsigned)f32_to_bf16(f.x) | ((unsigned)f32_to_bf16(f.y) << 16);
        p.y = (unsigned)f32_to_bf16(f.z) | ((unsigned)f32_to_bf16(f.w) << 16);
        ((uint2*)dst)[i] = p;
    }
}

// ---------------------------------------------------------------------------
// Kernel 2: LayerNorm over D_MODEL, emits c_bf16 (normalized) and h_bf16 (raw)
// ---------------------------------------------------------------------------
__global__ __launch_bounds__(256) void layernorm_cast_kernel(
    const float* __restrict__ h, const float* __restrict__ gamma,
    const float* __restrict__ beta,
    unsigned short* __restrict__ h_bf, unsigned short* __restrict__ c_bf) {
    int row = blockIdx.x;
    int t   = threadIdx.x;
    const float* hr = h + (size_t)row * DMODEL;
    float4 x = ((const float4*)hr)[t];

    __shared__ float red_s[256];
    __shared__ float red_q[256];
    red_s[t] = x.x + x.y + x.z + x.w;
    red_q[t] = x.x * x.x + x.y * x.y + x.z * x.z + x.w * x.w;
    __syncthreads();
    for (int off = 128; off > 0; off >>= 1) {
        if (t < off) { red_s[t] += red_s[t + off]; red_q[t] += red_q[t + off]; }
        __syncthreads();
    }
    float mu   = red_s[0] * (1.0f / DMODEL);
    float var  = red_q[0] * (1.0f / DMODEL) - mu * mu;
    float rinv = rsqrtf(var + 1e-5f);

    float4 g = ((const float4*)gamma)[t];
    float4 b = ((const float4*)beta)[t];
    float c0 = (x.x - mu) * rinv * g.x + b.x;
    float c1 = (x.y - mu) * rinv * g.y + b.y;
    float c2 = (x.z - mu) * rinv * g.z + b.z;
    float c3 = (x.w - mu) * rinv * g.w + b.w;

    uint2 pc, ph;
    pc.x = (unsigned)f32_to_bf16(c0) | ((unsigned)f32_to_bf16(c1) << 16);
    pc.y = (unsigned)f32_to_bf16(c2) | ((unsigned)f32_to_bf16(c3) << 16);
    ph.x = (unsigned)f32_to_bf16(x.x) | ((unsigned)f32_to_bf16(x.y) << 16);
    ph.y = (unsigned)f32_to_bf16(x.z) | ((unsigned)f32_to_bf16(x.w) << 16);
    ((uint2*)(c_bf + (size_t)row * DMODEL))[t] = pc;
    ((uint2*)(h_bf + (size_t)row * DMODEL))[t] = ph;
}

// ---------------------------------------------------------------------------
// Kernel 3: generic bf16 WMMA GEMM:  Out[M,N] = A[M,K] @ B[K,N] (row-major)
// Block tile 64x256, BK=32; 8 waves as 2(M) x 4(N); wave tile 32x64 =
// 2x4 accumulators -> 8 v_wmma per barrier pair. A tile DMA'd by the TDM,
// B tile K-pair-packed into LDS dwords (read in contiguous k order: lane half h holds k = 16h+s).
// MODE 0: store bf16.  MODE 1: store f32 with residual add (out = resid+acc).
// ---------------------------------------------------------------------------
template <int MODE>
__global__ __launch_bounds__(256) void gemm_bf16_kernel(
    const unsigned short* __restrict__ A, const unsigned short* __restrict__ B,
    int ldb, void* __restrict__ Out, int ldo,
    const float* __restrict__ Resid, int M, int N, int K) {
    __shared__ alignas(16) unsigned short As[64 * 32];  // 4 KB, row-major
    __shared__ unsigned int Bp[16 * 256];               // 16 KB, K-pair packed

    const int t  = threadIdx.x;
    const int w  = t >> 5;
    const int l  = t & 31;
    const int hk = l >> 4;
    const int ml = l & 15;
    const int wm = w >> 2;          // 0..1  (M)
    const int wn = w & 3;           // 0..3  (N)
    const int m0 = blockIdx.y * 64;
    const int n0 = blockIdx.x * 256;

    v8f zero = {};
    v8f acc[2][4];
#pragma unroll
    for (int mi = 0; mi < 2; ++mi)
#pragma unroll
        for (int ni = 0; ni < 4; ++ni) acc[mi][ni] = zero;

    const unsigned* As32 = (const unsigned*)As;
    const int nkt = K / 32;

    for (int kt = 0; kt < nkt; ++kt) {
#ifdef HAVE_TDM
        if (w == 0)  // one wave issues the A-tile DMA (64 rows x 32 bf16)
            tdm_load_tile_bf16(&As[0], &A[(size_t)m0 * K + kt * 32], 64, 32, K);
#else
        {   // manual A load: thread -> 8 bf16 (16B)
            int row = t >> 2, seg = t & 3;
            *(uint4*)&As[row * 32 + seg * 8] =
                *(const uint4*)&A[(size_t)(m0 + row) * K + kt * 32 + seg * 8];
        }
#endif
        // B: 32 rows x 256 cols; pack K-pairs (k even/odd) into dwords
        for (int task = t; task < 512; task += 256) {
            int k2 = task >> 5, ng = task & 31, n8 = ng * 8;
            const unsigned short* bsrc =
                &B[(size_t)(kt * 32 + 2 * k2) * ldb + n0 + n8];
            union { uint4 q; unsigned short s[8]; } r0, r1;
            r0.q = *(const uint4*)bsrc;
            r1.q = *(const uint4*)(bsrc + ldb);
            unsigned* dst = &Bp[k2 * 256 + n8];
#pragma unroll
            for (int i = 0; i < 8; ++i)
                dst[i] = (unsigned)r0.s[i] | ((unsigned)r1.s[i] << 16);
        }
        if (kt + 1 < nkt && (t & 7) == 0) {  // GL2 prefetch of next B tile
            __builtin_prefetch(&B[(size_t)((kt + 1) * 32 + (t >> 4)) * ldb + n0], 0, 0);
        }
        TDM_WAIT();
        __syncthreads();

        // --- fragments + 8 WMMAs ---
        FragBF af[2], bf[4];
#pragma unroll
        for (int mi = 0; mi < 2; ++mi) {
            const unsigned* p = &As32[(wm * 32 + mi * 16 + ml) * 16 + hk * 8];
            af[mi].q[0] = *(const uint4*)p;
            af[mi].q[1] = *(const uint4*)(p + 4);
        }
#pragma unroll
        for (int ni = 0; ni < 4; ++ni) {
            int n = wn * 64 + ni * 16 + ml;
#pragma unroll
            for (int v = 0; v < 8; ++v)
                bf[ni].u[v] = Bp[(hk * 8 + v) * 256 + n];
        }
#pragma unroll
        for (int mi = 0; mi < 2; ++mi)
#pragma unroll
            for (int ni = 0; ni < 4; ++ni)
                acc[mi][ni] = bf16_wmma(af[mi], bf[ni], acc[mi][ni]);
        __syncthreads();
    }

    // --- epilogue (C/D layout: lane l -> col l&15, row v + 8*(l>>4)) ---
#pragma unroll
    for (int mi = 0; mi < 2; ++mi)
#pragma unroll
        for (int ni = 0; ni < 4; ++ni)
#pragma unroll
            for (int v = 0; v < 8; ++v) {
                int grow = m0 + wm * 32 + mi * 16 + v + 8 * hk;
                int gcol = n0 + wn * 64 + ni * 16 + ml;
                size_t idx = (size_t)grow * ldo + gcol;
                if (MODE == 0) {
                    ((unsigned short*)Out)[idx] = f32_to_bf16(acc[mi][ni][v]);
                } else {
                    ((float*)Out)[idx] = Resid[idx] + acc[mi][ni][v];
                }
            }
}

// ---------------------------------------------------------------------------
// Kernel 4: fused attention (no softmax in reference!)
//   grid = (8 bn-groups of 4, 128 i-tiles of 16); block = 256 = 8 waves.
//   Waves (2k,2k+1) share head bn=g*4+k: each computes one 16-j half of the
//   score tile, then one 32-d half of S@V. Q/K tiles arrive via TDM; V tile
//   is VALU-packed into j-pair dwords, read in contiguous order (lane half h holds j = 16h+s).
// ---------------------------------------------------------------------------
__global__ __launch_bounds__(256) void attn_kernel(
    const unsigned short* __restrict__ Q,   // [4096][1024] bf16
    const unsigned short* __restrict__ Kb,  // [4096][1024] bf16
    const unsigned short* __restrict__ Vb,  // [4096][1024] bf16
    const int* __restrict__ mask,           // [2048][2]
    float* __restrict__ prob,               // [2048][2048][2][16] f32
    unsigned short* __restrict__ attn_vec)  // [4096][1024] bf16
{
    __shared__ alignas(16) unsigned short Qs[4][16][64];   // 8 KB
    __shared__ alignas(16) unsigned short Ks[4][32][64];   // 16 KB
    __shared__ unsigned int Vp[4][16][64];                 // 16 KB (j-pair packed)
    __shared__ alignas(16) unsigned short Sb[4][16][32];   // 4 KB  (bf16 scores)
    __shared__ alignas(16) float S32v[16][32][4];          // 8 KB  ([i][j][bnl])
    __shared__ int mj[32];

    const int g  = blockIdx.x;            // bn group: bns = g*4 .. g*4+3
    const int i0 = blockIdx.y * 16;
    const int b  = g >> 2;                // batch index (same for whole group)
    const int t  = threadIdx.x;
    const int w  = t >> 5;
    const int l  = t & 31;
    const int hk = l >> 4;
    const int ml = l & 15;
    const int bnl = w >> 1;               // local head 0..3
    const int dh  = w & 1;                // j-half (scores) / d-half (S@V)
    const int bn  = g * 4 + bnl;
    const int nn  = bn & 15;

    // --- preload Q tile for all 4 heads: 4 x (16 x 64) bf16 ---
#ifdef HAVE_TDM
    if (dh == 0) {   // waves 0,2,4,6: one head tile each via the TDM
        int row = i0 * 2 + b;
        tdm_load_tile_bf16(&Qs[bnl][0][0],
                           &Q[(size_t)row * 1024 + nn * 64], 16, 64, 2048);
    }
    TDM_WAIT();
#else
    for (int task = t; task < 512; task += 256) {
        int tb  = task >> 7;
        int rem = task & 127;
        int i   = rem >> 3;
        int dg  = rem & 7;
        int tbn = g * 4 + tb;
        int row = (i0 + i) * 2 + (tbn >> 4);
        *(uint4*)&Qs[tb][i][dg * 8] =
            *(const uint4*)&Q[(size_t)row * 1024 + (tbn & 15) * 64 + dg * 8];
    }
#endif

    v8f zero = {};
    v8f o[2]; o[0] = zero; o[1] = zero;   // O: 16 i x 32 d (this wave's d-half)

    const unsigned* Qd = (const unsigned*)&Qs[bnl][0][0]; // [16][32] dwords
    const unsigned* Kd = (const unsigned*)&Ks[bnl][0][0]; // [32][32] dwords
    const unsigned* Sd = (const unsigned*)&Sb[bnl][0][0]; // [16][16] dwords

    for (int jc = 0; jc < SEQ / 32; ++jc) {
        const int j0 = jc * 32;
        __syncthreads();   // previous chunk's tile reads done

        // --- K tile: 4 x (32 x 64) bf16 ---
#ifdef HAVE_TDM
        if (dh == 0) {
            int row = j0 * 2 + b;
            tdm_load_tile_bf16(&Ks[bnl][0][0],
                               &Kb[(size_t)row * 1024 + nn * 64], 32, 64, 2048);
        }
#else
        for (int task = t; task < 1024; task += 256) {
            int tb = task >> 8; int rem = task & 255;
            int j = rem >> 3;  int dg = rem & 7;
            int tbn = g * 4 + tb;
            int row = (j0 + j) * 2 + (tbn >> 4);
            *(uint4*)&Ks[tb][j][dg * 8] =
                *(const uint4*)&Kb[(size_t)row * 1024 + (tbn & 15) * 64 + dg * 8];
        }
#endif
        // --- V tile, packing (j, j+1) bf16 pairs per dword (B layout) ---
        for (int task = t; task < 512; task += 256) {
            int tb = task >> 7; int rem = task & 127;
            int j2 = rem >> 3;  int dg = rem & 7;
            int tbn = g * 4 + tb;
            int row0 = (j0 + 2 * j2) * 2 + (tbn >> 4);
            union { uint4 q; unsigned short s[8]; } v0, v1;
            v0.q = *(const uint4*)&Vb[(size_t)row0 * 1024 + (tbn & 15) * 64 + dg * 8];
            v1.q = *(const uint4*)&Vb[(size_t)(row0 + 2) * 1024 + (tbn & 15) * 64 + dg * 8];
#pragma unroll
            for (int i = 0; i < 8; ++i)
                Vp[tb][j2][dg * 8 + i] = (unsigned)v0.s[i] | ((unsigned)v1.s[i] << 16);
        }
        if (t < 32) mj[t] = mask[(size_t)(j0 + t) * 2 + b];
        TDM_WAIT();
        __syncthreads();

        // --- scores: S(16 x 16) for this wave's j-half; K accum over d=64 ---
        v8f s = zero;
#pragma unroll
        for (int ks = 0; ks < 2; ++ks) {
            const int dbase = ks * 32;
            FragBF qa, kf;
            const unsigned* p = &Qd[ml * 32 + (dbase >> 1) + hk * 8];
            qa.q[0] = *(const uint4*)p;
            qa.q[1] = *(const uint4*)(p + 4);
            const unsigned* pk = &Kd[(dh * 16 + ml) * 32 + (dbase >> 1) + hk * 8];
            kf.q[0] = *(const uint4*)pk;
            kf.q[1] = *(const uint4*)(pk + 4);
            s = bf16_wmma(qa, kf, s);
        }
        // scale + mask, stage f32 (for attn_prob) and bf16 (for S@V)
#pragma unroll
        for (int v = 0; v < 8; ++v) {
            int i    = v + 8 * hk;
            int jloc = dh * 16 + ml;
            float val = s[v] * SCALE_F;
            if (mj[jloc] == 1) val = MASK_VAL;
            S32v[i][jloc][bnl] = val;
            Sb[bnl][i][jloc] = f32_to_bf16(val);
        }
        __syncthreads();

        // --- attn_prob: [i][j][b][n]; 4 consecutive n per vec4, NT stores
        //     (537 MB streaming output — keep it out of L2) ---
        for (int task = t; task < 512; task += 256) {
            int i = task >> 5, j = task & 31;
            size_t addr = ((size_t)(i0 + i) * SEQ + (j0 + j)) * (BATCH * NHEAD) + g * 4;
            f32x4 sv = *(const f32x4*)&S32v[i][j][0];
            __builtin_nontemporal_store(sv, (f32x4*)&prob[addr]);
        }

        // --- O += S(16x32) @ V(32 x 32-d-half): 2 WMMAs ---
        FragBF sa;
        const unsigned* ps = &Sd[ml * 16 + hk * 8];
        sa.q[0] = *(const uint4*)ps;
        sa.q[1] = *(const uint4*)(ps + 4);
#pragma unroll
        for (int ni = 0; ni < 2; ++ni) {
            int d = dh * 32 + ni * 16 + ml;
            FragBF vf;
#pragma unroll
            for (int v = 0; v < 8; ++v)
                vf.u[v] = Vp[bnl][hk * 8 + v][d];
            o[ni] = bf16_wmma(sa, vf, o[ni]);
        }
    }

    // --- write attn_vec (bf16): row = (i0+i)*2+b, col = n*64+d ---
#pragma unroll
    for (int ni = 0; ni < 2; ++ni)
#pragma unroll
        for (int v = 0; v < 8; ++v) {
            int i = v + 8 * hk;
            int d = dh * 32 + ni * 16 + ml;
            int row = (i0 + i) * 2 + b;
            attn_vec[(size_t)row * 1024 + nn * 64 + d] = f32_to_bf16(o[ni][v]);
        }
}

// ---------------------------------------------------------------------------
// Host-side orchestration
// ---------------------------------------------------------------------------
extern "C" void kernel_launch(void* const* d_in, const int* in_sizes, int n_in,
                              void* d_out, int out_size, void* d_ws, size_t ws_size,
                              hipStream_t stream) {
    const float* h     = (const float*)d_in[0];
    const float* Wq    = (const float*)d_in[1];
    const float* Wkv   = (const float*)d_in[2];
    const float* Wo    = (const float*)d_in[3];
    const float* gamma = (const float*)d_in[4];
    const float* beta  = (const float*)d_in[5];
    const int*   amask = (const int*)d_in[6];

    // workspace layout (bf16 buffers), ~56 MB total
    char* ws = (char*)d_ws;
    unsigned short* h_bf   = (unsigned short*)ws;  ws += (size_t)ROWS * DMODEL * 2;
    unsigned short* c_bf   = (unsigned short*)ws;  ws += (size_t)ROWS * DMODEL * 2;
    unsigned short* Wq_bf  = (unsigned short*)ws;  ws += (size_t)DMODEL * DMODEL * 2;
    unsigned short* Wkv_bf = (unsigned short*)ws;  ws += (size_t)DMODEL * 2 * DMODEL * 2;
    unsigned short* Wo_bf  = (unsigned short*)ws;  ws += (size_t)DMODEL * DMODEL * 2;
    unsigned short* Qb     = (unsigned short*)ws;  ws += (size_t)ROWS * DMODEL * 2;
    unsigned short* Kbuf   = (unsigned short*)ws;  ws += (size_t)ROWS * DMODEL * 2;
    unsigned short* Vbuf   = (unsigned short*)ws;  ws += (size_t)ROWS * DMODEL * 2;
    unsigned short* AVbuf  = (unsigned short*)ws;  ws += (size_t)ROWS * DMODEL * 2;
    (void)in_sizes; (void)n_in; (void)out_size; (void)ws_size;

    float* out_f32  = (float*)d_out;                          // [2048,2,1024]
    float* out_prob = out_f32 + (size_t)SEQ * BATCH * DMODEL; // [2048,2048,2,16]

    // 1) weight casts
    {
        int n4 = DMODEL * DMODEL / 4;
        cast_bf16_kernel<<<(n4 + 255) / 256, 256, 0, stream>>>(Wq, Wq_bf, n4);
        int n4kv = DMODEL * 2 * DMODEL / 4;
        cast_bf16_kernel<<<(n4kv + 255) / 256, 256, 0, stream>>>(Wkv, Wkv_bf, n4kv);
        cast_bf16_kernel<<<(n4 + 255) / 256, 256, 0, stream>>>(Wo, Wo_bf, n4);
    }
    // 2) layernorm + activation casts
    layernorm_cast_kernel<<<ROWS, 256, 0, stream>>>(h, gamma, beta, h_bf, c_bf);

    // 3) projections (bf16 WMMA GEMMs), block tile 64x256
    dim3 ggrid(DMODEL / 256, ROWS / 64);
    gemm_bf16_kernel<0><<<ggrid, 256, 0, stream>>>(
        h_bf, Wq_bf, DMODEL, Qb, DMODEL, nullptr, ROWS, DMODEL, DMODEL);
    gemm_bf16_kernel<0><<<ggrid, 256, 0, stream>>>(
        c_bf, Wkv_bf, 2 * DMODEL, Kbuf, DMODEL, nullptr, ROWS, DMODEL, DMODEL);
    gemm_bf16_kernel<0><<<ggrid, 256, 0, stream>>>(
        c_bf, Wkv_bf + DMODEL, 2 * DMODEL, Vbuf, DMODEL, nullptr, ROWS, DMODEL, DMODEL);

    // 4) fused attention (scores -> attn_prob, S@V -> attn_vec)
    attn_kernel<<<dim3(8, SEQ / 16), 256, 0, stream>>>(
        Qb, Kbuf, Vbuf, amask, out_prob, AVbuf);

    // 5) output projection + residual: out = h + attn_vec @ Wo
    gemm_bf16_kernel<1><<<ggrid, 256, 0, stream>>>(
        AVbuf, Wo_bf, DMODEL, out_f32, DMODEL, h, ROWS, DMODEL, DMODEL);
}
